// LMHAttention_46643344834750
// MI455X (gfx1250) — compile-verified, matrix operand order checked by reading, not run on hardware
//
#include <hip/hip_runtime.h>

typedef _Float16 half_t;
typedef __attribute__((ext_vector_type(4)))  _Float16 v4h;
typedef __attribute__((ext_vector_type(8)))  _Float16 v8h;
typedef __attribute__((ext_vector_type(16))) _Float16 v16h;
typedef __attribute__((ext_vector_type(8)))  float    v8f;

#define WMMA_F16(a, b, c) \
  __builtin_amdgcn_wmma_f32_16x16x32_f16(false, (a), false, (b), (short)0, (c), false, false)

// Build a 16x32 f16 A/B fragment for one lane from a row-major K-run.
// ISA layout: lane = M%16 (or N%16); g = lane/16 selects K-halves:
//   VGPR0-3 <- K = 8g..8g+7, VGPR4-7 <- K = 16+8g..16+8g+7
__device__ __forceinline__ v16h frag_k32(const half_t* rowk0, int g) {
  v8h lo = *(const v8h*)(rowk0 + 8 * g);
  v8h hi = *(const v8h*)(rowk0 + 16 + 8 * g);
  return __builtin_shufflevector(lo, hi, 0, 1, 2, 3, 4, 5, 6, 7, 8, 9, 10, 11, 12, 13, 14, 15);
}

__device__ __forceinline__ v8f zero8() {
  v8f z = {0.f, 0.f, 0.f, 0.f, 0.f, 0.f, 0.f, 0.f};
  return z;
}

// CDNA5 async global->LDS copy (16B per lane), tracked by ASYNCcnt.
__device__ __forceinline__ void async_copy_b128(half_t* lds_dst, const half_t* gsrc) {
  unsigned lds = (unsigned)(uintptr_t)lds_dst;
  unsigned long long ga = (unsigned long long)(uintptr_t)gsrc;
  asm volatile("global_load_async_to_lds_b128 %0, %1, off" ::"v"(lds), "v"(ga) : "memory");
}
__device__ __forceinline__ void wait_async0() {
  asm volatile("s_wait_asynccnt 0x0" ::: "memory");
}

// ---------------- weight prep (fp32 -> f16, N-major layouts) ----------------

// Wq/Wk/Wv (H, DM, D) -> WT (H*D, DM): WT[n*1024+k] = W[(h*1024+k)*64+d]
__global__ void pack_headw(const float* __restrict__ W, half_t* __restrict__ WT) {
  int idx = blockIdx.x * 256 + threadIdx.x;      // 1M
  int n = idx >> 10, k = idx & 1023;
  int h = n >> 6, d = n & 63;
  WT[idx] = (half_t)W[((size_t)h * 1024 + k) * 64 + d];
}

// Wo (DM, DM) -> WoT (n, k) = Wo[k, n]
__global__ void pack_wo(const float* __restrict__ W, half_t* __restrict__ WT) {
  int idx = blockIdx.x * 256 + threadIdx.x;      // 1M
  int n = idx >> 10, k = idx & 1023;
  WT[idx] = (half_t)W[(size_t)k * 1024 + n];
}

// WE/WF (L, DK) -> (DK, L)
__global__ void pack_ef(const float* __restrict__ W, half_t* __restrict__ WT) {
  int idx = blockIdx.x * 256 + threadIdx.x;      // 1M
  int kk = idx >> 12, l = idx & 4095;
  WT[idx] = (half_t)W[(size_t)l * 256 + kk];
}

// ---------------- kernel A: Q/K/V projection GEMM (M=16384, N=1024, K=1024) --
// Double-buffered: async B copies + A global loads fly during WMMA compute.

__global__ __launch_bounds__(256, 1) void proj_gemm(
    const float* __restrict__ X,    // (16384, 1024) fp32
    const half_t* __restrict__ WT,  // (1024 n, 1024 k) f16 N-major
    const float* __restrict__ bias, // (1024)
    const int* __restrict__ mask,   // (B*L) int or nullptr
    half_t* __restrict__ out,       // f16
    int transposed)                 // 0: (b,h,l,d)   1: (b,h,d,l)
{
  constexpr int AS = 40;  // padded K-stride (halves), 80B = 5x16B
  __shared__ __align__(16) half_t Asm[2][128 * AS];
  __shared__ __align__(16) half_t Bsm[2][128 * AS];
  const int tid = threadIdx.x;
  const int wid = tid >> 5, lane = tid & 31;
  const int ln16 = lane & 15, g = lane >> 4;
  const int m0 = blockIdx.y * 128;
  const int n0 = blockIdx.x * 128;
  const int wm = (wid & 3) * 32;   // 4 M-groups of 32
  const int wn = (wid >> 2) * 64;  // 2 N-groups of 64

  v8f acc[2][4];
#pragma unroll
  for (int i = 0; i < 2; ++i)
#pragma unroll
    for (int j = 0; j < 4; ++j) acc[i][j] = zero8();

  float4 areg[4];

  auto stageB = [&](int buf, int k0) {
#pragma unroll
    for (int it = 0; it < 2; ++it) {
      int t = tid + it * 256;              // 0..511
      int row = t >> 2, seg = (t & 3) * 8;
      async_copy_b128(&Bsm[buf][row * AS + seg], WT + (size_t)(n0 + row) * 1024 + k0 + seg);
    }
  };
  auto loadA = [&](int k0) {
#pragma unroll
    for (int it = 0; it < 4; ++it) {
      int t = tid + it * 256;              // 0..1023
      int row = t >> 3, kq = (t & 7) * 4;
      areg[it] = *(const float4*)(X + (size_t)(m0 + row) * 1024 + k0 + kq);
    }
  };
  auto flushA = [&](int buf) {
#pragma unroll
    for (int it = 0; it < 4; ++it) {
      int t = tid + it * 256;
      int row = t >> 3, kq = (t & 7) * 4;
      const float4 v = areg[it];
      v4h hq; hq[0] = (half_t)v.x; hq[1] = (half_t)v.y; hq[2] = (half_t)v.z; hq[3] = (half_t)v.w;
      *(v4h*)(&Asm[buf][row * AS + kq]) = hq;
    }
  };

  // prologue: tile 0 fully staged
  stageB(0, 0);
  loadA(0);
  flushA(0);

  for (int kt = 0; kt < 32; ++kt) {
    const int cur = kt & 1, nxt = cur ^ 1;
    wait_async0();     // current tile's async copies (only ones outstanding)
    __syncthreads();   // current tile visible to all waves
    if (kt < 31) {
      stageB(nxt, (kt + 1) * 32);  // overlap with compute below
      loadA((kt + 1) * 32);        // f32 loads fly during compute
    }

    v16h af[2], bf[4];
#pragma unroll
    for (int mf = 0; mf < 2; ++mf) af[mf] = frag_k32(&Asm[cur][(wm + mf * 16 + ln16) * AS], g);
#pragma unroll
    for (int nf = 0; nf < 4; ++nf) bf[nf] = frag_k32(&Bsm[cur][(wn + nf * 16 + ln16) * AS], g);
#pragma unroll
    for (int mf = 0; mf < 2; ++mf)
#pragma unroll
      for (int nf = 0; nf < 4; ++nf) acc[mf][nf] = WMMA_F16(af[mf], bf[nf], acc[mf][nf]);

    if (kt < 31) flushA(nxt);      // convert+store next A tile after compute
    __syncthreads();               // readers of cur done before it is rewritten
  }

  // epilogue: bias, optional row-mask zero, store f16 in (b,h,l,d) or (b,h,d,l)
#pragma unroll
  for (int mf = 0; mf < 2; ++mf) {
#pragma unroll
    for (int i = 0; i < 8; ++i) {
      int mg = m0 + wm + mf * 16 + g * 8 + i;
      int b = mg >> 12, l = mg & 4095;
      int msk = mask ? mask[mg] : 0;
#pragma unroll
      for (int nf = 0; nf < 4; ++nf) {
        int ng = n0 + wn + nf * 16 + ln16;
        float val = acc[mf][nf][i] + bias[ng];
        if (msk) val = 0.f;
        int h = ng >> 6, d = ng & 63;
        size_t o = transposed ? (((size_t)(b * 16 + h) * 64 + d) * 4096 + l)
                              : (((size_t)(b * 16 + h) * 4096 + l) * 64 + d);
        out[o] = (half_t)val;
      }
    }
  }
}

// ------------- kernel B1: Kp^T = WE^T @ K   (per bh: M=256, N=64, K=4096) ----

__global__ __launch_bounds__(256, 1) void ef_proj_kp(
    const half_t* __restrict__ WEt,  // (256, 4096)
    const half_t* __restrict__ Kt,   // (BH, 64, 4096)
    const float* __restrict__ bE,    // (256)
    half_t* __restrict__ Kpt)        // (BH, 256, 64)
{
  constexpr int AS = 40;
  __shared__ __align__(16) half_t Asm[2][256 * AS];
  __shared__ __align__(16) half_t Bsm[2][64 * AS];
  const int bh = blockIdx.x;
  const half_t* Kb = Kt + (size_t)bh * 64 * 4096;
  const int tid = threadIdx.x, wid = tid >> 5, lane = tid & 31;
  const int ln16 = lane & 15, g = lane >> 4;
  const int wm = wid * 32;

  v8f acc[2][4];
#pragma unroll
  for (int i = 0; i < 2; ++i)
#pragma unroll
    for (int j = 0; j < 4; ++j) acc[i][j] = zero8();

  auto stage = [&](int buf, int k0) {
#pragma unroll
    for (int it = 0; it < 4; ++it) {
      int t = tid + it * 256;
      int row = t >> 2, seg = (t & 3) * 8;
      async_copy_b128(&Asm[buf][row * AS + seg], WEt + (size_t)row * 4096 + k0 + seg);
    }
    {
      int row = tid >> 2, seg = (tid & 3) * 8;
      async_copy_b128(&Bsm[buf][row * AS + seg], Kb + (size_t)row * 4096 + k0 + seg);
    }
  };

  stage(0, 0);
  for (int kt = 0; kt < 128; ++kt) {
    const int cur = kt & 1, nxt = cur ^ 1;
    wait_async0();
    __syncthreads();
    if (kt < 127) stage(nxt, (kt + 1) * 32);

    v16h af[2], bf[4];
#pragma unroll
    for (int mf = 0; mf < 2; ++mf) af[mf] = frag_k32(&Asm[cur][(wm + mf * 16 + ln16) * AS], g);
#pragma unroll
    for (int nf = 0; nf < 4; ++nf) bf[nf] = frag_k32(&Bsm[cur][(nf * 16 + ln16) * AS], g);
#pragma unroll
    for (int mf = 0; mf < 2; ++mf)
#pragma unroll
      for (int nf = 0; nf < 4; ++nf) acc[mf][nf] = WMMA_F16(af[mf], bf[nf], acc[mf][nf]);
    __syncthreads();
  }
#pragma unroll
  for (int mf = 0; mf < 2; ++mf)
#pragma unroll
    for (int i = 0; i < 8; ++i) {
      int row = wm + mf * 16 + g * 8 + i;  // kk
#pragma unroll
      for (int nf = 0; nf < 4; ++nf) {
        int col = nf * 16 + ln16;          // d
        Kpt[(size_t)bh * 16384 + (size_t)row * 64 + col] = (half_t)(acc[mf][nf][i] + bE[row]);
      }
    }
}

// ------------- kernel B2: Vp = V^T @ WF   (per bh: M=64, N=256, K=4096) ------

__global__ __launch_bounds__(256, 1) void ef_proj_vp(
    const half_t* __restrict__ Vt,   // (BH, 64, 4096)
    const half_t* __restrict__ WFt,  // (256, 4096)
    const float* __restrict__ bF,    // (256)
    half_t* __restrict__ Vp)         // (BH, 64, 256)
{
  constexpr int AS = 40;
  __shared__ __align__(16) half_t Asm[2][64 * AS];
  __shared__ __align__(16) half_t Bsm[2][256 * AS];
  const int bh = blockIdx.x;
  const half_t* Vb = Vt + (size_t)bh * 64 * 4096;
  const int tid = threadIdx.x, wid = tid >> 5, lane = tid & 31;
  const int ln16 = lane & 15, g = lane >> 4;
  const int wn = wid * 32;

  v8f acc[4][2];
#pragma unroll
  for (int i = 0; i < 4; ++i)
#pragma unroll
    for (int j = 0; j < 2; ++j) acc[i][j] = zero8();

  auto stage = [&](int buf, int k0) {
    {
      int row = tid >> 2, seg = (tid & 3) * 8;
      async_copy_b128(&Asm[buf][row * AS + seg], Vb + (size_t)row * 4096 + k0 + seg);
    }
#pragma unroll
    for (int it = 0; it < 4; ++it) {
      int t = tid + it * 256;
      int row = t >> 2, seg = (t & 3) * 8;
      async_copy_b128(&Bsm[buf][row * AS + seg], WFt + (size_t)row * 4096 + k0 + seg);
    }
  };

  stage(0, 0);
  for (int kt = 0; kt < 128; ++kt) {
    const int cur = kt & 1, nxt = cur ^ 1;
    wait_async0();
    __syncthreads();
    if (kt < 127) stage(nxt, (kt + 1) * 32);

    v16h af[4], bf[2];
#pragma unroll
    for (int mf = 0; mf < 4; ++mf) af[mf] = frag_k32(&Asm[cur][(mf * 16 + ln16) * AS], g);
#pragma unroll
    for (int nf = 0; nf < 2; ++nf) bf[nf] = frag_k32(&Bsm[cur][(wn + nf * 16 + ln16) * AS], g);
#pragma unroll
    for (int mf = 0; mf < 4; ++mf)
#pragma unroll
      for (int nf = 0; nf < 2; ++nf) acc[mf][nf] = WMMA_F16(af[mf], bf[nf], acc[mf][nf]);
    __syncthreads();
  }
#pragma unroll
  for (int mf = 0; mf < 4; ++mf)
#pragma unroll
    for (int i = 0; i < 8; ++i) {
      int row = mf * 16 + g * 8 + i;       // d
#pragma unroll
      for (int nf = 0; nf < 2; ++nf) {
        int col = wn + nf * 16 + ln16;     // kk
        Vp[(size_t)bh * 16384 + (size_t)row * 256 + col] = (half_t)(acc[mf][nf][i] + bF[col]);
      }
    }
}

// ------------- kernel C: scores -> softmax -> weights out + weights @ Vp^T ---

__global__ __launch_bounds__(256, 1) void attn_core(
    const half_t* __restrict__ Qh,   // (BH, 4096, 64)
    const half_t* __restrict__ Kpt,  // (BH, 256, 64)
    const half_t* __restrict__ Vp,   // (BH, 64, 256)
    float* __restrict__ Wout,        // (BH, 4096, 256) fp32 softmax weights
    half_t* __restrict__ O)          // (B, 4096, 1024) f16
{
  constexpr int KS = 72;   // Kp row stride (halves)
  constexpr int VS = 264;  // Vp row stride
  constexpr int CS = 40;   // per-wave weight-chunk stride
  __shared__ __align__(16) half_t smem[256 * KS + 8 * 16 * CS];  // 47,104 B
  half_t* KV = smem;                 // phase1: Kp^T(256x64); phase3: Vp(64x256)
  half_t* CH = smem + 256 * KS;      // per-wave 16x32 f16 weight chunks

  const int bh = blockIdx.y;
  const int b = bh >> 4, h = bh & 15;
  const int l0 = blockIdx.x * 128;
  const int tid = threadIdx.x, wid = tid >> 5, lane = tid & 31;
  const int ln16 = lane & 15, g = lane >> 4;

  const half_t* Qb = Qh + (size_t)bh * 4096 * 64;
  const half_t* Kpb = Kpt + (size_t)bh * 16384;
  const half_t* Vpb = Vp + (size_t)bh * 16384;

  // stage Kp^T cols 0..31 only (256 x 32) via async global->LDS
#pragma unroll
  for (int it = 0; it < 4; ++it) {
    int t = tid + it * 256;
    int row = t >> 2, seg = (t & 3) * 8;
    async_copy_b128(KV + row * KS + seg, Kpb + row * 64 + seg);
  }
  // Q A-fragments straight from global (row-major l,d)
  v16h aq[2];
  {
    const half_t* qrow = Qb + (size_t)(l0 + wid * 16 + ln16) * 64;
#pragma unroll
    for (int ks = 0; ks < 2; ++ks) aq[ks] = frag_k32(qrow + ks * 32, g);
  }
  wait_async0();
  __syncthreads();

  // scores: 16 rows x 256 cols per wave
  v8f acc[16];
#pragma unroll
  for (int nf = 0; nf < 16; ++nf) acc[nf] = zero8();
#pragma unroll
  for (int nf = 0; nf < 16; ++nf) {
#pragma unroll
    for (int ks = 0; ks < 2; ++ks) {
      v16h bfr = frag_k32(KV + (nf * 16 + ln16) * KS + ks * 32, g);
      acc[nf] = WMMA_F16(aq[ks], bfr, acc[nf]);
    }
  }
  __syncthreads();  // everyone done with Kp region

  // stage Vp (64 x 256) into the same region, async; overlaps with softmax
#pragma unroll
  for (int it = 0; it < 8; ++it) {
    int t = tid + it * 256;
    int row = t >> 5, seg = (t & 31) * 8;
    async_copy_b128(KV + row * VS + seg, Vpb + row * 256 + seg);
  }

  // exact softmax over DK=256, fully in registers (rows live in half-waves)
#pragma unroll
  for (int i = 0; i < 8; ++i) {
    float m = -3.0e38f;
#pragma unroll
    for (int nf = 0; nf < 16; ++nf) {
      float s = acc[nf][i] * 0.125f;  // 1/sqrt(64)
      acc[nf][i] = s;
      m = fmaxf(m, s);
    }
    m = fmaxf(m, __shfl_xor(m, 1, 16));
    m = fmaxf(m, __shfl_xor(m, 2, 16));
    m = fmaxf(m, __shfl_xor(m, 4, 16));
    m = fmaxf(m, __shfl_xor(m, 8, 16));
    float s = 0.f;
#pragma unroll
    for (int nf = 0; nf < 16; ++nf) {
      float e = __expf(acc[nf][i] - m);
      acc[nf][i] = e;
      s += e;
    }
    s += __shfl_xor(s, 1, 16);
    s += __shfl_xor(s, 2, 16);
    s += __shfl_xor(s, 4, 16);
    s += __shfl_xor(s, 8, 16);
    float inv = 1.0f / s;
#pragma unroll
    for (int nf = 0; nf < 16; ++nf) acc[nf][i] *= inv;
  }

  // weights (output #2) fp32
  {
    float* wb = Wout + ((size_t)bh * 4096 + l0 + wid * 16) * 256;
#pragma unroll
    for (int i = 0; i < 8; ++i) {
      int M = g * 8 + i;
#pragma unroll
      for (int nf = 0; nf < 16; ++nf) wb[(size_t)M * 256 + nf * 16 + ln16] = acc[nf][i];
    }
  }
  wait_async0();
  __syncthreads();  // Vp fully staged and visible

  // out = weights(16x256) @ Vp^T(256x64) : repack weights via per-wave LDS
  half_t* ch = CH + wid * 16 * CS;
  v8f oacc[4];
#pragma unroll
  for (int nf = 0; nf < 4; ++nf) oacc[nf] = zero8();
#pragma unroll
  for (int ks = 0; ks < 8; ++ks) {
#pragma unroll
    for (int tt = 0; tt < 2; ++tt) {
      int nf = ks * 2 + tt;
#pragma unroll
      for (int i = 0; i < 8; ++i) {
        int M = g * 8 + i;
        ch[M * CS + tt * 16 + ln16] = (half_t)acc[nf][i];
      }
    }
    asm volatile("s_wait_dscnt 0" ::: "memory");  // per-wave LDS transpose visible
    v16h afr = frag_k32(ch + ln16 * CS, g);
#pragma unroll
    for (int nf = 0; nf < 4; ++nf) {
      v16h bfr = frag_k32(KV + (nf * 16 + ln16) * VS + ks * 32, g);
      oacc[nf] = WMMA_F16(afr, bfr, oacc[nf]);
    }
  }

  // O (b, l, h*64 + d) f16
#pragma unroll
  for (int i = 0; i < 8; ++i) {
    int l = l0 + wid * 16 + g * 8 + i;
    half_t* orow = O + ((size_t)b * 4096 + l) * 1024 + h * 64;
#pragma unroll
    for (int nf = 0; nf < 4; ++nf) orow[nf * 16 + ln16] = (half_t)oacc[nf][i];
  }
}

// ------------- kernel D: output projection (M=16384, N=1024, K=1024) --------

__global__ __launch_bounds__(256, 1) void out_proj(
    const half_t* __restrict__ O,    // (16384, 1024) f16
    const half_t* __restrict__ WoT,  // (1024 n, 1024 k) f16
    const float* __restrict__ bo,    // (1024)
    float* __restrict__ Y)           // (16384, 1024) fp32
{
  constexpr int AS = 40;
  __shared__ __align__(16) half_t Asm[2][128 * AS];
  __shared__ __align__(16) half_t Bsm[2][128 * AS];
  const int tid = threadIdx.x;
  const int wid = tid >> 5, lane = tid & 31;
  const int ln16 = lane & 15, g = lane >> 4;
  const int m0 = blockIdx.y * 128;
  const int n0 = blockIdx.x * 128;
  const int wm = (wid & 3) * 32;
  const int wn = (wid >> 2) * 64;

  v8f acc[2][4];
#pragma unroll
  for (int i = 0; i < 2; ++i)
#pragma unroll
    for (int j = 0; j < 4; ++j) acc[i][j] = zero8();

  auto stage = [&](int buf, int k0) {
#pragma unroll
    for (int it = 0; it < 2; ++it) {
      int t = tid + it * 256;
      int row = t >> 2, seg = (t & 3) * 8;
      async_copy_b128(&Asm[buf][row * AS + seg], O + (size_t)(m0 + row) * 1024 + k0 + seg);
    }
#pragma unroll
    for (int it = 0; it < 2; ++it) {
      int t = tid + it * 256;
      int row = t >> 2, seg = (t & 3) * 8;
      async_copy_b128(&Bsm[buf][row * AS + seg], WoT + (size_t)(n0 + row) * 1024 + k0 + seg);
    }
  };

  stage(0, 0);
  for (int kt = 0; kt < 32; ++kt) {
    const int cur = kt & 1, nxt = cur ^ 1;
    wait_async0();
    __syncthreads();
    if (kt < 31) stage(nxt, (kt + 1) * 32);

    v16h af[2], bf[4];
#pragma unroll
    for (int mf = 0; mf < 2; ++mf) af[mf] = frag_k32(&Asm[cur][(wm + mf * 16 + ln16) * AS], g);
#pragma unroll
    for (int nf = 0; nf < 4; ++nf) bf[nf] = frag_k32(&Bsm[cur][(wn + nf * 16 + ln16) * AS], g);
#pragma unroll
    for (int mf = 0; mf < 2; ++mf)
#pragma unroll
      for (int nf = 0; nf < 4; ++nf) acc[mf][nf] = WMMA_F16(af[mf], bf[nf], acc[mf][nf]);
    __syncthreads();
  }
#pragma unroll
  for (int mf = 0; mf < 2; ++mf)
#pragma unroll
    for (int i = 0; i < 8; ++i) {
      int mg = m0 + wm + mf * 16 + g * 8 + i;
#pragma unroll
      for (int nf = 0; nf < 4; ++nf) {
        int ng = n0 + wn + nf * 16 + ln16;
        Y[(size_t)mg * 1024 + ng] = acc[mf][nf][i] + bo[ng];
      }
    }
}

// ---------------------------------- launch ----------------------------------

extern "C" void kernel_launch(void* const* d_in, const int* in_sizes, int n_in,
                              void* d_out, int out_size, void* d_ws, size_t ws_size,
                              hipStream_t stream) {
  (void)in_sizes; (void)n_in; (void)out_size; (void)ws_size;
  const float* query = (const float*)d_in[0];
  const float* key   = (const float*)d_in[1];
  const float* value = (const float*)d_in[2];
  const int*   mask  = (const int*)d_in[3];
  const float* Wq = (const float*)d_in[4];
  const float* bq = (const float*)d_in[5];
  const float* Wk = (const float*)d_in[6];
  const float* bk = (const float*)d_in[7];
  const float* Wv = (const float*)d_in[8];
  const float* bv = (const float*)d_in[9];
  const float* WE = (const float*)d_in[10];
  const float* bE = (const float*)d_in[11];
  const float* WF = (const float*)d_in[12];
  const float* bF = (const float*)d_in[13];
  const float* Wo = (const float*)d_in[14];
  const float* bo = (const float*)d_in[15];

  char* ws = (char*)d_ws;
  const size_t MB = 1024 * 1024;
  half_t* WqT = (half_t*)(ws + 0 * MB);
  half_t* WkT = (half_t*)(ws + 2 * MB);
  half_t* WvT = (half_t*)(ws + 4 * MB);
  half_t* WoT = (half_t*)(ws + 6 * MB);
  half_t* WEt = (half_t*)(ws + 8 * MB);
  half_t* WFt = (half_t*)(ws + 10 * MB);
  half_t* Qh  = (half_t*)(ws + 12 * MB);   // 32 MB
  half_t* Kt  = (half_t*)(ws + 44 * MB);   // 32 MB
  half_t* Vt  = (half_t*)(ws + 76 * MB);   // 32 MB
  half_t* Kpt = (half_t*)(ws + 108 * MB);  // 2 MB
  half_t* Vpw = (half_t*)(ws + 110 * MB);  // 2 MB
  half_t* Oh  = (half_t*)(ws + 112 * MB);  // 32 MB

  float* Ymha = (float*)d_out;              // (B, L, DM)
  float* Wsm  = (float*)d_out + 16777216;   // (B, H, L, DK)

  // weight prep
  pack_headw<<<4096, 256, 0, stream>>>(Wq, WqT);
  pack_headw<<<4096, 256, 0, stream>>>(Wk, WkT);
  pack_headw<<<4096, 256, 0, stream>>>(Wv, WvT);
  pack_wo<<<4096, 256, 0, stream>>>(Wo, WoT);
  pack_ef<<<4096, 256, 0, stream>>>(WE, WEt);
  pack_ef<<<4096, 256, 0, stream>>>(WF, WFt);

  // Q/K/V projections (K/V transposed + masked)
  dim3 gA(8, 128);
  proj_gemm<<<gA, 256, 0, stream>>>(query, WqT, bq, nullptr, Qh, 0);
  proj_gemm<<<gA, 256, 0, stream>>>(key,   WkT, bk, mask,    Kt, 1);
  proj_gemm<<<gA, 256, 0, stream>>>(value, WvT, bv, mask,    Vt, 1);

  // Linformer E/F projections
  ef_proj_kp<<<64, 256, 0, stream>>>(WEt, Kt, bE, Kpt);
  ef_proj_vp<<<64, 256, 0, stream>>>(Vt, WFt, bF, Vpw);

  // attention core (scores, softmax, weights out, weights @ Vp^T)
  attn_core<<<dim3(32, 64), 256, 0, stream>>>(Qh, Kpt, Vpw, Wsm, Oh);

  // output projection
  out_proj<<<gA, 256, 0, stream>>>(Oh, WoT, bo, Ymha);
}
